// MultiheadSelfAttention_29850022708068
// MI455X (gfx1250) — compile-verified, matrix operand order checked by reading, not run on hardware
//
#include <hip/hip_runtime.h>
#include <hip/hip_bf16.h>
#include <math.h>

typedef __attribute__((ext_vector_type(16))) _Float16 v16h;
typedef __attribute__((ext_vector_type(8)))  float    v8f;
typedef unsigned int u32x4 __attribute__((ext_vector_type(4)));
typedef int          i32x4 __attribute__((ext_vector_type(4)));
typedef int          i32x8 __attribute__((ext_vector_type(8)));

#define H_DIM 2880
#define NH    64
#define NKV   8
#define DH    64
#define GROUP 8
#define T_SEQ 1024
#define B_SZ  2
#define BT    (B_SZ * T_SEQ)
#define QN    (NH * DH)      // 4096
#define KVN   (NKV * DH)     // 512
#define SCALE_ATT 0.125f     // 1/sqrt(64)
#define ROPE_THETA  150000.0f
#define ROPE_FACTOR 32.0f
#define BKK 32
#define BM  256
#define BN  64

// ---------------------------------------------------------------------------
// WMMA helpers (CDNA5 gfx1250, wave32)
// ---------------------------------------------------------------------------
__device__ __forceinline__ v8f wmma_f16(v16h a, v16h b, v8f c) {
  // D = A(16x32 f16) x B(32x16 f16) + C(16x16 f32)
  return __builtin_amdgcn_wmma_f32_16x16x32_f16(
      /*neg_a=*/false, a, /*neg_b=*/false, b,
      /*c_mod=*/(short)0, c, /*reuse_a=*/false, /*reuse_b=*/false);
}

// Load a 16x32 f16 operand fragment from a row-major [16][>=32] buffer.
// ISA layout (16-bit A/B operands): lanes 0-15 -> row r, K chunks [0..7],[16..23];
// lanes 16-31 -> row r, K chunks [8..15],[24..31]. Two 16-byte loads per lane.
__device__ __forceinline__ v16h frag_ld(const _Float16* p, int stride_halfs) {
  const int lane = threadIdx.x & 31;
  const int r  = lane & 15;
  const int hi = lane >> 4;
  const _Float16* q = p + (size_t)r * stride_halfs + hi * 8;
  union { v16h v; uint4 u[2]; } U;
  U.u[0] = *(const uint4*)(q);
  U.u[1] = *(const uint4*)(q + 16);
  return U.v;
}

// ---------------------------------------------------------------------------
// Tensor Data Mover: one 2D tile global->LDS per instruction (TENSOR_LOAD_TO_LDS,
// tracked by TENSORcnt). Descriptor per ISA 8.3/8.4:
//  group0: [1:0]=count=1, [63:32]=lds_addr, [120:64]=global_addr, [127:126]=type=2
//  group1: data_size=1 (2B), tensor_dim0/1, tile_dim0/1, tensor_dim0_stride
// Generic pointers to LDS keep the AS3 offset in the low 32 bits.
// ---------------------------------------------------------------------------
__device__ __forceinline__ unsigned lds_addr_of(const void* p) {
  return (unsigned)(unsigned long long)(size_t)p;
}

__device__ __forceinline__ void tdm_load_2d_f16(
    unsigned lds_off, const void* gaddr,
    unsigned tile_rows, unsigned tile_halfs,
    unsigned tensor_w, unsigned tensor_h,
    unsigned long long row_stride_halfs) {
  unsigned long long ga = (unsigned long long)(size_t)gaddr;
  u32x4 g0;
  g0[0] = 1u;                                              // count=1 (valid)
  g0[1] = lds_off;                                         // lds_addr (bytes)
  g0[2] = (unsigned)(ga & 0xFFFFFFFFu);                    // global_addr lo
  g0[3] = (unsigned)((ga >> 32) & 0x01FFFFFFu) | (2u << 30); // addr hi | type=2
  i32x8 g1;
  g1[0] = (int)(1u << 16);                                 // data_size=1 -> 2B
  g1[1] = (int)((tensor_w & 0xFFFFu) << 16);               // tensor_dim0[15:0]
  g1[2] = (int)(((tensor_w >> 16) & 0xFFFFu) |
                ((tensor_h & 0xFFFFu) << 16));             // td0[31:16]|td1[15:0]
  g1[3] = (int)(((tensor_h >> 16) & 0xFFFFu) |
                ((tile_halfs & 0xFFFFu) << 16));           // td1[31:16]|tile_dim0
  g1[4] = (int)(tile_rows & 0xFFFFu);                      // tile_dim1 (dim2=0)
  g1[5] = (int)(unsigned)(row_stride_halfs & 0xFFFFFFFFull); // dim0_stride lo
  g1[6] = (int)(unsigned)((row_stride_halfs >> 32) & 0xFFFFull);
  g1[7] = 0;
  i32x4 z4 = {0, 0, 0, 0};
#if __clang_major__ >= 23
  i32x8 z8 = {0, 0, 0, 0, 0, 0, 0, 0};
  __builtin_amdgcn_tensor_load_to_lds(g0, g1, z4, z4, z8, 0);
#else
  __builtin_amdgcn_tensor_load_to_lds(g0, g1, z4, z4, 0);
#endif
}

// ---------------------------------------------------------------------------
// Elementwise conversion / transpose kernels
// ---------------------------------------------------------------------------
__global__ void cvt_f32_f16(const float* __restrict__ src,
                            _Float16* __restrict__ dst, int n) {
  int i = blockIdx.x * blockDim.x + threadIdx.x;
  if (i < n) dst[i] = (_Float16)src[i];
}

// src [K][N] f32 -> dst [N][K] f16
__global__ void transpose_f32_f16(const float* __restrict__ src,
                                  _Float16* __restrict__ dst, int K, int N) {
  int idx = blockIdx.x * blockDim.x + threadIdx.x;
  if (idx >= K * N) return;
  int n = idx / K;
  int k = idx - n * K;
  dst[(size_t)n * K + k] = (_Float16)src[(size_t)k * N + n];
}

// ---------------------------------------------------------------------------
// Tiled WMMA GEMM: C[M][N] = A[M][K] (f16) * Bt[N][K]^T (f16) + bias[N], f32 out
// Block: 256 threads = 8 waves; tile 256x64; K-step 32; double-buffered LDS
// filled by the Tensor Data Mover (wave 0 issues 2 descriptors per tile).
// Wave w -> 64x32 sub-tile (4x2 of 16x16 WMMA frags -> 8 wmma per K-step).
// ---------------------------------------------------------------------------
__global__ __launch_bounds__(256) void gemm_f16f32(
    const _Float16* __restrict__ A,    // [M][K]
    const _Float16* __restrict__ Bt,   // [N][K]  (weight pre-transposed)
    const float* __restrict__ bias,    // [N]
    float* __restrict__ C,             // [M][N]
    int M, int N, int K) {
  __shared__ __align__(16) _Float16 As[2][BM * BKK];   // 2 x 16 KB
  __shared__ __align__(16) _Float16 Bs[2][BN * BKK];   // 2 x 4 KB

  const int tid  = threadIdx.x;
  const int wave = tid >> 5;
  const int lane = tid & 31;
  const int m0 = blockIdx.y * BM;
  const int n0 = blockIdx.x * BN;
  const int wr = wave >> 1;   // 0..3 -> 64-row band
  const int wc = wave & 1;    // 0..1 -> 32-col band

  const unsigned a_lds0 = lds_addr_of(&As[0][0]);
  const unsigned b_lds0 = lds_addr_of(&Bs[0][0]);
  const unsigned a_bufsz = BM * BKK * 2u;
  const unsigned b_bufsz = BN * BKK * 2u;

  // Stage one K-tile into LDS buffer `buf`: two TDM descriptors (A and B).
  auto stage = [&](int buf, int k0) {
    tdm_load_2d_f16(a_lds0 + (unsigned)buf * a_bufsz,
                    &A[(size_t)m0 * K + k0],
                    BM, BKK, (unsigned)K, (unsigned)M,
                    (unsigned long long)K);
    tdm_load_2d_f16(b_lds0 + (unsigned)buf * b_bufsz,
                    &Bt[(size_t)n0 * K + k0],
                    BN, BKK, (unsigned)K, (unsigned)N,
                    (unsigned long long)K);
  };

  v8f acc[4][2] = {};
  if (wave == 0) stage(0, 0);

  for (int k0 = 0; k0 < K; k0 += BKK) {
    const int cur = (k0 / BKK) & 1;
    const bool more = (k0 + BKK) < K;
    if (wave == 0) {
      if (more) {
        stage(cur ^ 1, k0 + BKK);                 // prefetch next tile
        __builtin_amdgcn_s_wait_tensorcnt(2);     // current tile's DMAs done
      } else {
        __builtin_amdgcn_s_wait_tensorcnt(0);
      }
    }
    __syncthreads();                // all waves: current tile visible

    const _Float16* Abuf = &As[cur][0];
    const _Float16* Bbuf = &Bs[cur][0];
    v16h af[4], bf[2];
    #pragma unroll
    for (int i = 0; i < 4; ++i)
      af[i] = frag_ld(&Abuf[(wr * 64 + i * 16) * BKK], BKK);
    #pragma unroll
    for (int j = 0; j < 2; ++j)
      bf[j] = frag_ld(&Bbuf[(wc * 32 + j * 16) * BKK], BKK);

    #pragma unroll
    for (int i = 0; i < 4; ++i)
      #pragma unroll
      for (int j = 0; j < 2; ++j)
        acc[i][j] = wmma_f16(af[i], bf[j], acc[i][j]);

    __syncthreads();                // all waves done reading before re-stage
  }

  // Epilogue: C/D frag layout: lane(0-15)->N=lane,M=vgpr; lane(16-31)->N=lane-16,M=vgpr+8
  const int r  = lane & 15;
  const int hi = lane >> 4;
  #pragma unroll
  for (int i = 0; i < 4; ++i) {
    #pragma unroll
    for (int j = 0; j < 2; ++j) {
      int nG = n0 + wc * 32 + j * 16 + r;
      float bvv = bias ? bias[nG] : 0.0f;
      #pragma unroll
      for (int e = 0; e < 8; ++e) {
        int mG = m0 + wr * 64 + i * 16 + e + 8 * hi;
        C[(size_t)mG * N + nG] = acc[i][j][e] + bvv;
      }
    }
  }
}

// ---------------------------------------------------------------------------
// RoPE: src f32 [BT][heads*DH] -> dst f16 (rotated). One thread per (bt,h,pair).
// ---------------------------------------------------------------------------
__global__ void rope_kernel(const float* __restrict__ src,
                            const int* __restrict__ pos,
                            _Float16* __restrict__ dst, int heads) {
  int idx = blockIdx.x * blockDim.x + threadIdx.x;
  int total = BT * heads * (DH / 2);
  if (idx >= total) return;
  int i   = idx & 31;          // pair index 0..31
  int tmp = idx >> 5;
  int h   = tmp % heads;
  int bt  = tmp / heads;
  float p   = (float)pos[bt] / ROPE_FACTOR;
  float inv = __powf(ROPE_THETA, -(float)(2 * i) / (float)DH);
  float ang = p * inv;
  float s, c;
  __sincosf(ang, &s, &c);
  size_t base = (size_t)bt * heads * DH + (size_t)h * DH;
  float x1 = src[base + 2 * i];
  float x2 = src[base + 2 * i + 1];
  dst[base + 2 * i]     = (_Float16)(x1 * c - x2 * s);
  dst[base + 2 * i + 1] = (_Float16)(x1 * s + x2 * c);
}

// ---------------------------------------------------------------------------
// Sliding-window attention with sink, WMMA for QK^T and P*V.
// Grid: (T/16, NKV*2, B). Block: 128 threads = 4 waves.
// Wave w handles query head = hkv*GROUP + hg*4 + w for query rows [t0, t0+16).
// Window (<=128) spans 9 key tiles of 16 -> 160 padded score columns.
// ---------------------------------------------------------------------------
__global__ __launch_bounds__(128) void attn_kernel(
    const _Float16* __restrict__ qh,   // [BT][QN]  (RoPE'd)
    const _Float16* __restrict__ kh,   // [BT][KVN] (RoPE'd)
    const _Float16* __restrict__ vh,   // [BT][KVN]
    const float* __restrict__ sink,    // [NH]
    const int* __restrict__ slide_flag,
    const int* __restrict__ slide_wnd,
    _Float16* __restrict__ yh)         // [BT][QN]
{
  const int qt  = blockIdx.x;
  const int hkv = blockIdx.y >> 1;
  const int hg  = blockIdx.y & 1;
  const int b   = blockIdx.z;
  const int t0  = qt * 16;
  const int wave = threadIdx.x >> 5;
  const int lane = threadIdx.x & 31;
  const int r  = lane & 15;
  const int hi = lane >> 4;
  const int head = hkv * GROUP + hg * 4 + wave;
  const int sliding = *slide_flag;
  const int wnd = *slide_wnd;

  const int NT = 9;     // key tiles covering [t0-128, t0+16)
  const int SW = 160;   // padded score columns (5 chunks of 32)
  const int ts_base = t0 - 128;

  __shared__ __align__(16) _Float16 vT[DH * SW];      // V^T window: [d][s]
  __shared__ __align__(16) float    sS[4 * 16 * SW];  // per-wave scores
  __shared__ __align__(16) _Float16 sP[4 * 16 * SW];  // per-wave probs

  // Stage V^T window cooperatively (shared across 4 heads in this block)
  for (int e = threadIdx.x; e < DH * SW; e += blockDim.x) {
    int d = e / SW, j = e - d * SW;
    int s = ts_base + j;
    _Float16 val = (_Float16)0.0f;
    if (s >= 0 && s < T_SEQ)
      val = vh[((size_t)(b * T_SEQ + s) * NKV + hkv) * DH + d];
    vT[d * SW + j] = val;
  }
  __syncthreads();

  // Q fragments: rows t0..t0+15, K-dim = head dim (two 32-wide chunks)
  const _Float16* qbase = qh + (size_t)(b * T_SEQ + t0) * QN + head * DH;
  v16h aq0 = frag_ld(qbase,      QN);
  v16h aq1 = frag_ld(qbase + 32, QN);

  float*    myS = &sS[wave * 16 * SW];
  _Float16* myP = &sP[wave * 16 * SW];

  // ---- scores: S = (Q K^T) * scale, masked ----
  for (int j = 0; j < NT; ++j) {
    int ts = ts_base + j * 16;
    int s  = ts + r;                                  // this lane's key column
    int sc = s < 0 ? 0 : (s >= T_SEQ ? T_SEQ - 1 : s);
    const _Float16* kr = kh + ((size_t)(b * T_SEQ + sc) * NKV + hkv) * DH;
    union { v16h v; uint4 u[2]; } B0, B1;
    B0.u[0] = *(const uint4*)(kr +      hi * 8);
    B0.u[1] = *(const uint4*)(kr + 16 + hi * 8);
    B1.u[0] = *(const uint4*)(kr + 32 + hi * 8);
    B1.u[1] = *(const uint4*)(kr + 48 + hi * 8);

    v8f sc8 = {};
    sc8 = wmma_f16(aq0, B0.v, sc8);
    sc8 = wmma_f16(aq1, B1.v, sc8);

    #pragma unroll
    for (int e = 0; e < 8; ++e) {
      int m = e + 8 * hi;       // query row within tile
      int t = t0 + m;
      bool ok = (s >= 0) && (s <= t) && (!sliding || (t - s) < wnd);
      myS[m * SW + j * 16 + r] = ok ? sc8[e] * SCALE_ATT : -1.0e30f;
    }
  }
  __syncthreads();

  // ---- softmax with sink (row r; lane pair {r, r+16} splits 144 cols) ----
  {
    float snk = sink[head];
    int c0 = hi * 72, c1 = c0 + 72;
    float mx = -3.0e38f;
    for (int c = c0; c < c1; ++c) mx = fmaxf(mx, myS[r * SW + c]);
    float mo = __shfl_xor(mx, 16);
    mx = fmaxf(fmaxf(mx, mo), snk);
    float sum = 0.0f;
    for (int c = c0; c < c1; ++c) sum += __expf(myS[r * SW + c] - mx);
    sum += __shfl_xor(sum, 16);
    float denom = sum + __expf(snk - mx);   // sink column (dropped after softmax)
    float invd = 1.0f / denom;
    for (int c = c0; c < c1; ++c)
      myP[r * SW + c] = (_Float16)(__expf(myS[r * SW + c] - mx) * invd);
    if (hi) {
      for (int c = 144; c < 160; ++c) myP[r * SW + c] = (_Float16)0.0f;
    }
  }
  __syncthreads();

  // ---- y = P @ V : M=16, K=160 (5 chunks), N=64 (4 tiles) ----
  v8f out[4] = {};
  #pragma unroll
  for (int kc = 0; kc < 5; ++kc) {
    v16h ap = frag_ld(myP + kc * 32, SW);
    #pragma unroll
    for (int nt = 0; nt < 4; ++nt) {
      v16h bvf = frag_ld(&vT[(nt * 16) * SW + kc * 32], SW);
      out[nt] = wmma_f16(ap, bvf, out[nt]);
    }
  }

  // store y (f16) at [b][t][head*DH + d]
  #pragma unroll
  for (int nt = 0; nt < 4; ++nt) {
    #pragma unroll
    for (int e = 0; e < 8; ++e) {
      int t = t0 + e + 8 * hi;
      int d = nt * 16 + r;
      yh[(size_t)(b * T_SEQ + t) * QN + head * DH + d] = (_Float16)out[nt][e];
    }
  }
}

// ---------------------------------------------------------------------------
// Launcher
// ---------------------------------------------------------------------------
extern "C" void kernel_launch(void* const* d_in, const int* in_sizes, int n_in,
                              void* d_out, int out_size, void* d_ws, size_t ws_size,
                              hipStream_t stream) {
  (void)in_sizes; (void)n_in; (void)out_size; (void)ws_size;

  const float* x        = (const float*)d_in[0];
  const int*   positions= (const int*)d_in[1];
  // d_in[2] causal_mask: implemented analytically, unused
  const float* Wq = (const float*)d_in[3];
  const float* bq = (const float*)d_in[4];
  const float* Wk = (const float*)d_in[5];
  const float* bk = (const float*)d_in[6];
  const float* Wv = (const float*)d_in[7];
  const float* bv = (const float*)d_in[8];
  const float* Wo = (const float*)d_in[9];
  const float* bo = (const float*)d_in[10];
  const float* sink  = (const float*)d_in[11];
  const int*   sflag = (const int*)d_in[12];
  const int*   swnd  = (const int*)d_in[13];
  float* out = (float*)d_out;

  char* ws = (char*)d_ws;
  size_t off = 0;
  auto alloc = [&](size_t bytes) -> void* {
    void* p = ws + off;
    off = (off + bytes + 255) & ~(size_t)255;
    return p;
  };
  _Float16* xh  = (_Float16*)alloc((size_t)BT * H_DIM * 2);
  _Float16* WqT = (_Float16*)alloc((size_t)QN * H_DIM * 2);
  _Float16* WkT = (_Float16*)alloc((size_t)KVN * H_DIM * 2);
  _Float16* WvT = (_Float16*)alloc((size_t)KVN * H_DIM * 2);
  _Float16* WoT = (_Float16*)alloc((size_t)H_DIM * QN * 2);
  float*    qf  = (float*)alloc((size_t)BT * QN * 4);
  float*    kf  = (float*)alloc((size_t)BT * KVN * 4);
  float*    vf  = (float*)alloc((size_t)BT * KVN * 4);
  _Float16* qrh = (_Float16*)alloc((size_t)BT * QN * 2);
  _Float16* krh = (_Float16*)alloc((size_t)BT * KVN * 2);
  _Float16* vhh = (_Float16*)alloc((size_t)BT * KVN * 2);
  _Float16* yh  = (_Float16*)alloc((size_t)BT * QN * 2);

  int n;
  n = BT * H_DIM;
  cvt_f32_f16<<<(n + 255) / 256, 256, 0, stream>>>(x, xh, n);
  n = H_DIM * QN;
  transpose_f32_f16<<<(n + 255) / 256, 256, 0, stream>>>(Wq, WqT, H_DIM, QN);
  n = H_DIM * KVN;
  transpose_f32_f16<<<(n + 255) / 256, 256, 0, stream>>>(Wk, WkT, H_DIM, KVN);
  transpose_f32_f16<<<(n + 255) / 256, 256, 0, stream>>>(Wv, WvT, H_DIM, KVN);
  n = QN * H_DIM;
  transpose_f32_f16<<<(n + 255) / 256, 256, 0, stream>>>(Wo, WoT, QN, H_DIM);

  dim3 gq(QN / BN, BT / BM);
  gemm_f16f32<<<gq, 256, 0, stream>>>(xh, WqT, bq, qf, BT, QN, H_DIM);
  dim3 gk(KVN / BN, BT / BM);
  gemm_f16f32<<<gk, 256, 0, stream>>>(xh, WkT, bk, kf, BT, KVN, H_DIM);
  gemm_f16f32<<<gk, 256, 0, stream>>>(xh, WvT, bv, vf, BT, KVN, H_DIM);

  n = BT * NH * (DH / 2);
  rope_kernel<<<(n + 255) / 256, 256, 0, stream>>>(qf, positions, qrh, NH);
  n = BT * NKV * (DH / 2);
  rope_kernel<<<(n + 255) / 256, 256, 0, stream>>>(kf, positions, krh, NKV);
  n = BT * KVN;
  cvt_f32_f16<<<(n + 255) / 256, 256, 0, stream>>>(vf, vhh, n);

  dim3 ga(T_SEQ / 16, NKV * 2, B_SZ);
  attn_kernel<<<ga, 128, 0, stream>>>(qrh, krh, vhh, sink, sflag, swnd, yh);

  dim3 go(H_DIM / BN, BT / BM);
  gemm_f16f32<<<go, 256, 0, stream>>>(yh, WoT, bo, out, BT, H_DIM, QN);
}
